// HybridQLSTM_65481071396755
// MI455X (gfx1250) — matrix operand order repaired and checked by reading, not compiled, not run on hardware
//
#include <hip/hip_runtime.h>
#include <hip/hip_bf16.h>

// Problem dims
#define B_  64
#define S_  512
#define E_  256
#define H_  256
#define T_  64
#define G4  1024   // 4*H

typedef __attribute__((ext_vector_type(16))) __bf16       v16bf;
typedef __attribute__((ext_vector_type(8)))  float        v8f;
typedef __attribute__((ext_vector_type(4)))  unsigned int u32x4;
typedef __attribute__((ext_vector_type(4)))  float        f32x4;

union FragBF { v16bf v; u32x4 q[2]; };

// ---- WMMA fragment loaders (layouts per CDNA5 ISA 7.12.2) ----
// A 16x32 bf16: lane holds row m=lane%16; elems 0-7 -> K = kb + half*8 + e,
//               elems 8-15 -> K = kb + 16 + half*8 + e.
__device__ __forceinline__ v16bf load_a_frag(const __hip_bfloat16* A, int lda,
                                             int row, int kb, int half) {
  const __hip_bfloat16* p = A + (size_t)row * lda + kb + half * 8;
  FragBF f;
  f.q[0] = *(const u32x4*)(p);
  f.q[1] = *(const u32x4*)(p + 16);
  return f.v;
}
// B 32x16 bf16 (computing A @ W^T, W row-major [N,K]):
// lane holds col n=lane%16; elems 0-7 -> K = kb + half*8 + e, elems 8-15 -> K = kb + 16 + half*8 + e.
__device__ __forceinline__ v16bf load_b_frag(const __hip_bfloat16* W, int ldw,
                                             int col, int kb, int half) {
  const __hip_bfloat16* p = W + (size_t)col * ldw + kb + half * 8;
  FragBF f;
  f.q[0] = *(const u32x4*)(p);
  f.q[1] = *(const u32x4*)(p + 16);
  return f.v;
}

__device__ __forceinline__ float sigmoidf_(float x) {
  return 1.0f / (1.0f + __expf(-x));
}

// =====================================================================
// Generic bf16 WMMA GEMM: C[M,N] = A[M,K] @ W[N,K]^T + bias[N]  (f32 out)
// One wave computes a 16x64 tile (1 M-tile x 4 N-tiles, A-frag reuse x4).
// K is compile-time; the k-loop fully unrolls and is software-pipelined:
// fragments for step k+1 are issued before the WMMAs of step k, so the
// scheduler can use partial s_wait_loadcnt instead of full drains.
// Requires M%16==0, N%64==0, KC%32==0.
// =====================================================================
template <int KC>
__global__ void gemm_bf16_wmma(const __hip_bfloat16* __restrict__ A,
                               const __hip_bfloat16* __restrict__ W,
                               const float* __restrict__ bias,
                               float* __restrict__ C,
                               int M, int N) {
  int wave = (blockIdx.x * blockDim.x + threadIdx.x) >> 5;
  int lane = threadIdx.x & 31;
  int lm = lane & 15, half = lane >> 4;
  int nGroups = N >> 6;
  int mTile = wave / nGroups;
  int nGroup = wave - mTile * nGroups;
  if (mTile * 16 >= M) return;
  int mBase = mTile * 16;
  int nBase = nGroup * 64;

  v8f acc[4] = {};
  // pipeline prologue
  v16bf fa = load_a_frag(A, KC, mBase + lm, 0, half);
  v16bf fb[4];
#pragma unroll
  for (int nt = 0; nt < 4; ++nt)
    fb[nt] = load_b_frag(W, KC, nBase + nt * 16 + lm, 0, half);

#pragma unroll
  for (int kb = 0; kb < KC; kb += 32) {
    v16bf na = fa;
    v16bf nb[4] = {fb[0], fb[1], fb[2], fb[3]};
    if (kb + 32 < KC) {  // statically resolved per unrolled iteration
      na = load_a_frag(A, KC, mBase + lm, kb + 32, half);
#pragma unroll
      for (int nt = 0; nt < 4; ++nt)
        nb[nt] = load_b_frag(W, KC, nBase + nt * 16 + lm, kb + 32, half);
    }
#pragma unroll
    for (int nt = 0; nt < 4; ++nt)
      acc[nt] = __builtin_amdgcn_wmma_f32_16x16x32_bf16(
          false, fa, false, fb[nt], (short)0, acc[nt], false, false);
    fa = na;
#pragma unroll
    for (int nt = 0; nt < 4; ++nt) fb[nt] = nb[nt];
  }

#pragma unroll
  for (int nt = 0; nt < 4; ++nt) {
    int col = nBase + nt * 16 + lm;
    float bv = bias[col];
#pragma unroll
    for (int r = 0; r < 8; ++r) {
      int row = mBase + half * 8 + r;
      C[(size_t)row * N + col] = acc[nt][r] + bv;
    }
  }
}

// =====================================================================
// One LSTM recurrent step:
//   gates[64,1024] = xpre + Hin @ Wh^T ; cell update ; Hout = h (bf16)
// Wave w owns tile (b: mTile*16.., h: nTile*16..) and computes the four
// gate tiles (weight row blocks 0..3) so the cell update is local.
// Same one-deep register prefetch pipeline as the GEMM (K=256 static).
// Launch <<<8, 256>>> = 64 waves.
// =====================================================================
template <int GI, int GF, int GG, int GO>
__global__ void lstm_step_wmma(const __hip_bfloat16* __restrict__ Hin,  // [64,256]
                               const __hip_bfloat16* __restrict__ Wh,   // [1024,256]
                               const float* __restrict__ xpre,          // [64,1024]
                               float* __restrict__ Cst,                 // [64,256]
                               __hip_bfloat16* __restrict__ Hout) {     // [64,256]
  int wave = (blockIdx.x * blockDim.x + threadIdx.x) >> 5;
  int lane = threadIdx.x & 31;
  int lm = lane & 15, half = lane >> 4;
  int mTile = wave & 3;    // 4 tiles over B=64
  int nTile = wave >> 2;   // 16 tiles over H=256
  int mBase = mTile * 16, nBase = nTile * 16;

  v8f acc[4] = {};
  v16bf fa = load_a_frag(Hin, H_, mBase + lm, 0, half);
  v16bf fb[4];
#pragma unroll
  for (int g = 0; g < 4; ++g)
    fb[g] = load_b_frag(Wh, H_, g * H_ + nBase + lm, 0, half);

#pragma unroll
  for (int kb = 0; kb < H_; kb += 32) {
    v16bf na = fa;
    v16bf nb[4] = {fb[0], fb[1], fb[2], fb[3]};
    if (kb + 32 < H_) {
      na = load_a_frag(Hin, H_, mBase + lm, kb + 32, half);
#pragma unroll
      for (int g = 0; g < 4; ++g)
        nb[g] = load_b_frag(Wh, H_, g * H_ + nBase + lm, kb + 32, half);
    }
#pragma unroll
    for (int g = 0; g < 4; ++g)
      acc[g] = __builtin_amdgcn_wmma_f32_16x16x32_bf16(
          false, fa, false, fb[g], (short)0, acc[g], false, false);
    fa = na;
#pragma unroll
    for (int g = 0; g < 4; ++g) fb[g] = nb[g];
  }

  int col = nBase + lm;
#pragma unroll
  for (int r = 0; r < 8; ++r) {
    int row = mBase + half * 8 + r;  // batch index
    const float* xp = xpre + (size_t)row * G4;
    float gi = acc[GI][r] + xp[GI * H_ + col];
    float gf = acc[GF][r] + xp[GF * H_ + col];
    float gg = acc[GG][r] + xp[GG * H_ + col];
    float go = acc[GO][r] + xp[GO * H_ + col];
    float iv = sigmoidf_(gi);
    float fv = sigmoidf_(gf);
    float gv = tanhf(gg);
    float ov = sigmoidf_(go);
    size_t idx = (size_t)row * H_ + col;
    float c = fv * Cst[idx] + iv * gv;
    Cst[idx] = c;
    Hout[idx] = __float2bfloat16(ov * tanhf(c));
  }
}

// =====================================================================
// Tag projection + log_softmax over T=64.
// One wave per 16 (s,b)-rows; 4 N-tiles cover all 64 tag columns, so the
// softmax reduction is in-register + __shfl_xor over the 16-lane half
// (masks 1,2,4,8 never cross halves -> rows stay separate).
// Launch <<<256, 256>>> = 2048 waves.
// =====================================================================
__global__ void tag_wmma(const __hip_bfloat16* __restrict__ A,   // [S*B, 256]
                         const __hip_bfloat16* __restrict__ Wt,  // [64, 256]
                         const float* __restrict__ bt,           // [64]
                         float* __restrict__ out) {               // [B,S,64]
  int wave = (blockIdx.x * blockDim.x + threadIdx.x) >> 5;
  int lane = threadIdx.x & 31;
  int lm = lane & 15, half = lane >> 4;
  int mBase = wave * 16;
  if (mBase >= S_ * B_) return;

  v8f acc[4] = {};
  v16bf fa = load_a_frag(A, H_, mBase + lm, 0, half);
  v16bf fb[4];
#pragma unroll
  for (int nt = 0; nt < 4; ++nt)
    fb[nt] = load_b_frag(Wt, H_, nt * 16 + lm, 0, half);

#pragma unroll
  for (int kb = 0; kb < H_; kb += 32) {
    v16bf na = fa;
    v16bf nb[4] = {fb[0], fb[1], fb[2], fb[3]};
    if (kb + 32 < H_) {
      na = load_a_frag(A, H_, mBase + lm, kb + 32, half);
#pragma unroll
      for (int nt = 0; nt < 4; ++nt)
        nb[nt] = load_b_frag(Wt, H_, nt * 16 + lm, kb + 32, half);
    }
#pragma unroll
    for (int nt = 0; nt < 4; ++nt)
      acc[nt] = __builtin_amdgcn_wmma_f32_16x16x32_bf16(
          false, fa, false, fb[nt], (short)0, acc[nt], false, false);
    fa = na;
#pragma unroll
    for (int nt = 0; nt < 4; ++nt) fb[nt] = nb[nt];
  }

  float bv[4];
#pragma unroll
  for (int nt = 0; nt < 4; ++nt) bv[nt] = bt[nt * 16 + lm];

#pragma unroll
  for (int r = 0; r < 8; ++r) {
    float v0 = acc[0][r] + bv[0];
    float v1 = acc[1][r] + bv[1];
    float v2 = acc[2][r] + bv[2];
    float v3 = acc[3][r] + bv[3];
    float mx = fmaxf(fmaxf(v0, v1), fmaxf(v2, v3));
#pragma unroll
    for (int msk = 1; msk <= 8; msk <<= 1)
      mx = fmaxf(mx, __shfl_xor(mx, msk, 32));
    float se = __expf(v0 - mx) + __expf(v1 - mx) +
               __expf(v2 - mx) + __expf(v3 - mx);
#pragma unroll
    for (int msk = 1; msk <= 8; msk <<= 1)
      se += __shfl_xor(se, msk, 32);
    float lse = mx + __logf(se);
    int row = mBase + half * 8 + r;    // row = s*B + b
    int s = row >> 6;                  // /64
    int b = row & 63;
    float* o = out + ((size_t)b * S_ + s) * T_;
    o[0 * 16 + lm] = v0 - lse;
    o[1 * 16 + lm] = v1 - lse;
    o[2 * 16 + lm] = v2 - lse;
    o[3 * 16 + lm] = v3 - lse;
  }
}

// ---- Prep kernels ----
__global__ void gather_embed(const int* __restrict__ sent,   // [B,S]
                             const float* __restrict__ emb,  // [V,E]
                             __hip_bfloat16* __restrict__ Ebf) {  // [S,B,E]
  size_t i = (size_t)blockIdx.x * blockDim.x + threadIdx.x;
  const size_t total = (size_t)S_ * B_ * (E_ / 4);
  if (i >= total) return;
  int e4 = (int)(i % (E_ / 4));
  size_t sb = i / (E_ / 4);
  int b = (int)(sb % B_);
  int s = (int)(sb / B_);
  int tok = sent[(size_t)b * S_ + s];
  f32x4 f = *(const f32x4*)(emb + (size_t)tok * E_ + e4 * 4);
  __hip_bfloat16* d = Ebf + sb * E_ + e4 * 4;
  d[0] = __float2bfloat16(f.x);
  d[1] = __float2bfloat16(f.y);
  d[2] = __float2bfloat16(f.z);
  d[3] = __float2bfloat16(f.w);
}

__global__ void convert_w(const float* __restrict__ src, int srcStride, int colOff,
                          int rows, int cols, __hip_bfloat16* __restrict__ dst) {
  int i = blockIdx.x * blockDim.x + threadIdx.x;
  if (i >= rows * cols) return;
  int r = i / cols, c = i - r * cols;
  dst[i] = __float2bfloat16(src[(size_t)r * srcStride + colOff + c]);
}

__global__ void prep_misc(const float* bf, const float* bi, const float* bu,
                          const float* bo, const float* bih, const float* bhh,
                          float* bg, float* bsum, float* C1, float* C2,
                          __hip_bfloat16* H0) {
  int i = blockIdx.x * blockDim.x + threadIdx.x;
  if (i < G4) {
    int g = i >> 8, r = i & 255;
    const float* p = (g == 0) ? bf : (g == 1) ? bi : (g == 2) ? bu : bo;
    bg[i] = p[r];
    bsum[i] = bih[i] + bhh[i];
  }
  if (i < B_ * H_) {
    C1[i] = 0.0f;
    C2[i] = 0.0f;
    H0[i] = __float2bfloat16(0.0f);
  }
}

// =====================================================================
extern "C" void kernel_launch(void* const* d_in, const int* in_sizes, int n_in,
                              void* d_out, int out_size, void* d_ws, size_t ws_size,
                              hipStream_t stream) {
  const int*   sent = (const int*)d_in[0];
  const float* emb  = (const float*)d_in[1];
  const float* Wf   = (const float*)d_in[2];
  const float* bf   = (const float*)d_in[3];
  const float* Wi   = (const float*)d_in[4];
  const float* bi   = (const float*)d_in[5];
  const float* Wu   = (const float*)d_in[6];
  const float* bu   = (const float*)d_in[7];
  const float* Wo   = (const float*)d_in[8];
  const float* bo   = (const float*)d_in[9];
  const float* Wih  = (const float*)d_in[10];
  const float* Whh  = (const float*)d_in[11];
  const float* bih  = (const float*)d_in[12];
  const float* bhh  = (const float*)d_in[13];
  const float* Wtag = (const float*)d_in[14];
  const float* btag = (const float*)d_in[15];
  float* out = (float*)d_out;

  char* ws = (char*)d_ws;
  size_t off = 0;
  auto alloc = [&](size_t bytes) -> char* {
    char* p = ws + off;
    off += (bytes + 255) & ~(size_t)255;
    return p;
  };
  __hip_bfloat16* Wgx    = (__hip_bfloat16*)alloc((size_t)G4 * E_ * 2);  // [1024,256]
  __hip_bfloat16* Wgh    = (__hip_bfloat16*)alloc((size_t)G4 * H_ * 2);  // [1024,256]
  __hip_bfloat16* WihB   = (__hip_bfloat16*)alloc((size_t)G4 * H_ * 2);
  __hip_bfloat16* WhhB   = (__hip_bfloat16*)alloc((size_t)G4 * H_ * 2);
  __hip_bfloat16* WtB    = (__hip_bfloat16*)alloc((size_t)T_ * H_ * 2);
  float*          bg     = (float*)alloc((size_t)G4 * 4);
  float*          bsum   = (float*)alloc((size_t)G4 * 4);
  float*          C1     = (float*)alloc((size_t)B_ * H_ * 4);
  float*          C2     = (float*)alloc((size_t)B_ * H_ * 4);
  __hip_bfloat16* H0     = (__hip_bfloat16*)alloc((size_t)B_ * H_ * 2);
  __hip_bfloat16* Ebf    = (__hip_bfloat16*)alloc((size_t)S_ * B_ * E_ * 2);   // 16 MB
  __hip_bfloat16* gateOut= (__hip_bfloat16*)alloc((size_t)S_ * B_ * H_ * 2);   // 16 MB
  __hip_bfloat16* lstmOut= (__hip_bfloat16*)alloc((size_t)S_ * B_ * H_ * 2);   // 16 MB
  float*          Xpre   = (float*)alloc((size_t)S_ * B_ * G4 * 4);            // 128 MB (reused A&C)

  // ---- weight/bias prep ----
  const float* Wg4[4] = {Wf, Wi, Wu, Wo};
  for (int g = 0; g < 4; ++g) {
    convert_w<<<(H_ * E_ + 255) / 256, 256, 0, stream>>>(
        Wg4[g], E_ + H_, 0, H_, E_, Wgx + (size_t)g * H_ * E_);
    convert_w<<<(H_ * H_ + 255) / 256, 256, 0, stream>>>(
        Wg4[g], E_ + H_, E_, H_, H_, Wgh + (size_t)g * H_ * H_);
  }
  convert_w<<<(G4 * H_ + 255) / 256, 256, 0, stream>>>(Wih, H_, 0, G4, H_, WihB);
  convert_w<<<(G4 * H_ + 255) / 256, 256, 0, stream>>>(Whh, H_, 0, G4, H_, WhhB);
  convert_w<<<(T_ * H_ + 255) / 256, 256, 0, stream>>>(Wtag, H_, 0, T_, H_, WtB);
  prep_misc<<<(B_ * H_ + 255) / 256, 256, 0, stream>>>(bf, bi, bu, bo, bih, bhh,
                                                       bg, bsum, C1, C2, H0);
  gather_embed<<<(S_ * B_ * E_ / 4 + 255) / 256, 256, 0, stream>>>(sent, emb, Ebf);

  const int M = S_ * B_;                            // 32768
  const int gemmBlocks = (M / 16) * (G4 / 64) / 8;  // 4096 blocks (8 waves each)

  // ---- Phase A: Xpre = embeds @ Wgx^T + bg  (parallel WMMA GEMM) ----
  gemm_bf16_wmma<E_><<<gemmBlocks, 256, 0, stream>>>(Ebf, Wgx, bg, Xpre, M, G4);

  // ---- Phase B: QLSTM recurrence (gate order f,i,u,o -> i=1,f=0,g=2,o=3) ----
  for (int t = 0; t < S_; ++t) {
    const __hip_bfloat16* Hin =
        (t == 0) ? H0 : gateOut + (size_t)(t - 1) * B_ * H_;
    lstm_step_wmma<1, 0, 2, 3><<<8, 256, 0, stream>>>(
        Hin, Wgh, Xpre + (size_t)t * B_ * G4, C1,
        gateOut + (size_t)t * B_ * H_);
  }

  // ---- Phase C: Xpre = gate_out @ W_ih^T + (b_ih+b_hh)  (reuse buffer) ----
  gemm_bf16_wmma<H_><<<gemmBlocks, 256, 0, stream>>>(gateOut, WihB, bsum, Xpre, M, G4);

  // ---- Phase D: nn.LSTM recurrence (torch order i,f,g,o) ----
  for (int t = 0; t < S_; ++t) {
    const __hip_bfloat16* Hin =
        (t == 0) ? H0 : lstmOut + (size_t)(t - 1) * B_ * H_;
    lstm_step_wmma<0, 1, 2, 3><<<8, 256, 0, stream>>>(
        Hin, WhhB, Xpre + (size_t)t * B_ * G4, C2,
        lstmOut + (size_t)t * B_ * H_);
  }

  // ---- Phase E: tag logits + log_softmax ----
  tag_wmma<<<(M / 16) / 8, 256, 0, stream>>>(lstmOut, WtB, btag, out);
}
